// ConditionalAux_36412732735781
// MI455X (gfx1250) — matrix operand order repaired and checked by reading, not compiled, not run on hardware
//
#include <hip/hip_runtime.h>
#include <stdint.h>

#define B_    32
#define S_    512
#define COND_ 32
#define DFULL_ 1024
#define D_    992
#define EPS_  1e-9f
#define NLLW_ 0.01f

typedef __attribute__((ext_vector_type(16))) __bf16 v16bf;
typedef __attribute__((ext_vector_type(8)))  float  v8f;

__device__ __forceinline__ uint16_t f2bf(float f) {
  union { float f; uint32_t u; } c; c.f = f;
  uint32_t u = c.u;
  u += 0x7fffu + ((u >> 16) & 1u);          // round-to-nearest-even
  return (uint16_t)(u >> 16);
}
__device__ __forceinline__ float bf2f(uint16_t h) {
  union { uint32_t u; float f; } c; c.u = ((uint32_t)h) << 16;
  return c.f;
}

// ---------------- kernel 0: zero the atomic accumulators ----------------
__global__ void k_zero(float* acc, int n) {
  int t = threadIdx.x + blockIdx.x * blockDim.x;
  if (t < n) acc[t] = 0.f;
}

// ---------------- kernel 1: per-row softmax, A_sig/A_reg build, NLL -----
// one block (256 thr) per (b,i) row; S=512 -> 2 elements per thread
__global__ void k_rows(const float* __restrict__ qt0, const float* __restrict__ rate,
                       const float* __restrict__ xlog, const int* __restrict__ data,
                       const int* __restrict__ xt,
                       uint16_t* __restrict__ aSig, uint16_t* __restrict__ aReg,
                       float* __restrict__ denomArr, float* __restrict__ Zxt,
                       float* __restrict__ nllSum)
{
  __shared__ float red[256];
  int row = blockIdx.x;                 // 0 .. B*D-1
  int b = row / D_, i = row % D_;
  const float* x = xlog + ((size_t)b * DFULL_ + COND_ + i) * S_;
  int xti = xt[row];
  int dti = data[row];
  int t = threadIdx.x;

  float x0 = x[t], x1 = x[t + 256];
  red[t] = fmaxf(x0, x1); __syncthreads();
  for (int o = 128; o > 0; o >>= 1) { if (t < o) red[t] = fmaxf(red[t], red[t + o]); __syncthreads(); }
  float m = red[0]; __syncthreads();

  float e0 = __expf(x0 - m), e1 = __expf(x1 - m);
  red[t] = e0 + e1; __syncthreads();
  for (int o = 128; o > 0; o >>= 1) { if (t < o) red[t] += red[t + o]; __syncthreads(); }
  float sum = red[0];
  float inv = 1.0f / sum;

  size_t rowOff = (size_t)row * S_;
  #pragma unroll
  for (int p = 0; p < 2; ++p) {
    int s = t + p * 256;
    float ps   = (p ? e1 : e0) * inv;                          // p0t[b,i,s]
    float qcol = qt0 [((size_t)b * S_ + s) * S_ + xti];        // qt0[b,s,xt]
    float rcol = rate[((size_t)b * S_ + s) * S_ + xti];        // rate[b,s,xt]
    aSig[rowOff + s] = f2bf(ps / (qcol + EPS_));
    aReg[rowOff + s] = (s == xti) ? (uint16_t)0 : f2bf(rcol);  // mask folded in
  }
  if (t == 0) {
    denomArr[row] = qt0[((size_t)b * S_ + dti) * S_ + xti] + EPS_;
    Zxt[row]      = -rate[((size_t)b * S_ + xti) * S_ + xti];
    float logp = x[dti] - m - __logf(sum);
    atomicAdd(nllSum, logp);
  }
}

// ---------------- kernel 2: rate diagonal ----------------
__global__ void k_zrow(const float* __restrict__ rate, float* __restrict__ Zrow) {
  int b = blockIdx.x, s = threadIdx.x;  // block 512
  Zrow[b * S_ + s] = -rate[((size_t)b * S_ + s) * S_ + s];
}

// ---------------- kernel 3: baseZ[b] = sum_i Zxt[b,i] ----------------
__global__ void k_basez(const float* __restrict__ Zxt, float* __restrict__ baseZ) {
  __shared__ float red[256];
  int b = blockIdx.x, t = threadIdx.x;
  float s = 0.f;
  for (int i = t; i < D_; i += 256) s += Zxt[(size_t)b * D_ + i];
  red[t] = s; __syncthreads();
  for (int o = 128; o > 0; o >>= 1) { if (t < o) red[t] += red[t + o]; __syncthreads(); }
  if (t == 0) baseZ[b] = red[0];
}

// ---------------- kernel 4: swizzle qt0 into WMMA B-fragment order ------
// B (32x16 KxN) k order chosen here (same as A): lanes 0-15 hold K=0..15, lanes 16-31 hold K=16..31,
// N = lane&15. Tile = 512 bf16 halves, lane-major so each lane reads 32
// contiguous bytes in the GEMM hot loop.
__global__ void k_swz(const float* __restrict__ qt0,
                      uint32_t* __restrict__ qSw1,   // GEMM1: B[s][k] = qt0[b][k][s]
                      uint32_t* __restrict__ qSw2)   // GEMM2: B[s][k] = qt0[b][s][k]
{
  int b = blockIdx.y;
  int tile = blockIdx.x;                // 0..511
  int kt = tile >> 5, nt = tile & 31;   // K-chunk (16), N-tile (32)
  int t = threadIdx.x;                  // 256
  int lane = t >> 3, ep = t & 7;
  int N  = lane & 15;
  int Kg = kt * 32 + ((lane >> 4) << 4) + 2 * ep;
  int Ng = nt * 16 + N;
  size_t qb = (size_t)b * S_ * S_;
  float a1 = qt0[qb + (size_t)Ng * S_ + Kg];
  float c1 = qt0[qb + (size_t)Ng * S_ + Kg + 1];
  float a2 = qt0[qb + (size_t)Kg * S_ + Ng];
  float c2 = qt0[qb + (size_t)(Kg + 1) * S_ + Ng];
  size_t dstU = ((((size_t)b * 16 + kt) * 32 + nt) * 512 + (size_t)lane * 16 + 2 * ep) >> 1;
  qSw1[dstU] = (uint32_t)f2bf(a1) | ((uint32_t)f2bf(c1) << 16);
  qSw2[dstU] = (uint32_t)f2bf(a2) | ((uint32_t)f2bf(c2) << 16);
}

// ---------------- kernel 5: fused dual-GEMM + epilogue ----------------
// grid (31, B); 256 thr = 8 waves (2 M x 4 N). Each wave: 16 rows x 128 cols,
// both GEMMs, K=512 via 16 chunks of 32 (bf16 WMMA 16x16x32, fp32 acc).
__global__ void __launch_bounds__(256, 1) k_main(
    const float* __restrict__ qt0, const int* __restrict__ data,
    const uint16_t* __restrict__ aSig, const uint16_t* __restrict__ aReg,
    const uint32_t* __restrict__ qSw1, const uint32_t* __restrict__ qSw2,
    const float* __restrict__ denomArr, const float* __restrict__ Zxt,
    const float* __restrict__ Zrow, const float* __restrict__ baseZ,
    float* __restrict__ accReg, float* __restrict__ accOut, float* __restrict__ accNorm)
{
  int b = blockIdx.y;
  int t = threadIdx.x;
  int wave = t >> 5, lane = t & 31;
  int waveM = wave >> 2, waveN = wave & 3;
  int mt = blockIdx.x * 2 + waveM;      // 0..61 (D = 62*16)
  int mBase = mt * 16;
  int laneM = lane & 15;
  int hi = lane >> 4;

  v8f acc1[8], acc2[8];
  #pragma unroll
  for (int n = 0; n < 8; ++n) {
    #pragma unroll
    for (int e = 0; e < 8; ++e) { acc1[n][e] = 0.f; acc2[n][e] = 0.f; }
  }

  // A 16x32 bf16, k order chosen here (same as B): lane holds row M=lane&15; halves e=0..15 -> K=hi*16+e
  // => one contiguous 32B run in row-major storage.
  size_t aRow = ((size_t)b * D_ + mBase + laneM) * S_;
  const uint16_t* aSigRow = aSig + aRow;
  const uint16_t* aRegRow = aReg + aRow;
  size_t bTileBase = (size_t)b * 512;   // (b*16 kt)*32 nt tiles

  union U16 { uint4 q[2]; v16bf v; };

  for (int kt = 0; kt < 16; ++kt) {
    int kOff = kt * 32 + hi * 16;
    __builtin_prefetch(aSigRow + kOff + 32, 0, 1);  // next K-chunk
    U16 aS_, aR_;
    aS_.q[0] = *reinterpret_cast<const uint4*>(aSigRow + kOff);
    aS_.q[1] = *reinterpret_cast<const uint4*>(aSigRow + kOff + 8);
    aR_.q[0] = *reinterpret_cast<const uint4*>(aRegRow + kOff);
    aR_.q[1] = *reinterpret_cast<const uint4*>(aRegRow + kOff + 8);
    #pragma unroll
    for (int ns = 0; ns < 8; ++ns) {
      int nt = ns * 4 + waveN;
      size_t tu = ((bTileBase + (size_t)kt * 32 + nt) * 512 + (size_t)lane * 16) >> 1;
      U16 b1_, b2_;
      const uint4* p1 = reinterpret_cast<const uint4*>(qSw1 + tu);
      b1_.q[0] = p1[0]; b1_.q[1] = p1[1];
      const uint4* p2 = reinterpret_cast<const uint4*>(qSw2 + tu);
      b2_.q[0] = p2[0]; b2_.q[1] = p2[1];
      acc1[ns] = __builtin_amdgcn_wmma_f32_16x16x32_bf16(false, aR_.v, false, b1_.v,
                                                         (short)0, acc1[ns], false, false);
      acc2[ns] = __builtin_amdgcn_wmma_f32_16x16x32_bf16(false, aS_.v, false, b2_.v,
                                                         (short)0, acc2[ns], false, false);
    }
  }

  // ---- epilogue: C layout (f32 16x16): N = lane&15, M = v + hi*8 ----
  int   iv[8]; float dn[8]; float zx[8]; const float* qrow[8];
  #pragma unroll
  for (int v = 0; v < 8; ++v) {
    int i = mBase + v + hi * 8;
    iv[v] = i;
    size_t r = (size_t)b * D_ + i;
    dn[v] = denomArr[r];
    zx[v] = Zxt[r];
    qrow[v] = qt0 + ((size_t)b * S_ + data[r]) * S_;
  }
  float bz = baseZ[b];
  float regS = 0.f, outS = 0.f, nrmS = 0.f;
  #pragma unroll
  for (int ns = 0; ns < 8; ++ns) {
    int k = (ns * 4 + waveN) * 16 + laneM;
    float zk = Zrow[b * S_ + k];
    #pragma unroll
    for (int v = 0; v < 8; ++v) {
      size_t e = ((size_t)b * D_ + iv[v]) * S_ + k;
      float as = bf2f(aSig[e]);                 // p0t / (qt0[b,k,xt]+eps)
      float w0 = bf2f(aReg[e]);                 // mask * rate[b,k,xt]
      float c1 = acc1[ns][v];
      float c2 = acc2[ns][v];
      regS += as * c1;
      float outer = w0 * qrow[v][k] / dn[v];
      outS += outer * __logf(c2 + EPS_);
      nrmS += outer / (bz - zx[v] + zk);
    }
  }

  __shared__ float red[256 * 3];
  red[t] = regS; red[t + 256] = outS; red[t + 512] = nrmS;
  __syncthreads();
  for (int o = 128; o > 0; o >>= 1) {
    if (t < o) { red[t] += red[t + o]; red[t + 256] += red[t + 256 + o]; red[t + 512] += red[t + 512 + o]; }
    __syncthreads();
  }
  if (t == 0) {
    atomicAdd(&accReg[b],  red[0]);
    atomicAdd(&accOut[b],  red[256]);
    atomicAdd(&accNorm[b], red[512]);
  }
}

// ---------------- kernel 6: final scalar ----------------
__global__ void k_final(const float* accReg, const float* accOut, const float* accNorm,
                        const float* nllSum, float* out)
{
  __shared__ float red[32];
  int t = threadIdx.x;   // 32
  float v = -accOut[t] / accNorm[t] + accReg[t];
  red[t] = v; __syncthreads();
  for (int o = 16; o > 0; o >>= 1) { if (t < o) red[t] += red[t + o]; __syncthreads(); }
  if (t == 0) {
    float neg_elbo = red[0] / (float)B_;
    float nll = -nllSum[0] / (float)(B_ * D_);
    out[0] = neg_elbo + NLLW_ * nll;
  }
}

extern "C" void kernel_launch(void* const* d_in, const int* in_sizes, int n_in,
                              void* d_out, int out_size, void* d_ws, size_t ws_size,
                              hipStream_t stream)
{
  (void)in_sizes; (void)n_in; (void)out_size; (void)ws_size;
  const float* qt0  = (const float*)d_in[0];
  const float* rate = (const float*)d_in[1];
  const float* xlog = (const float*)d_in[2];
  const int*   data = (const int*)d_in[3];
  const int*   xt   = (const int*)d_in[4];
  float* out = (float*)d_out;

  char* ws = (char*)d_ws;
  size_t off = 0;
  auto alloc = [&](size_t bytes) -> char* {
    char* p = ws + off; off = (off + bytes + 255) & ~(size_t)255; return p;
  };
  const size_t NR = (size_t)B_ * D_;
  uint16_t* aSig   = (uint16_t*)alloc(NR * S_ * 2);            // 32.5 MB
  uint16_t* aReg   = (uint16_t*)alloc(NR * S_ * 2);            // 32.5 MB
  uint32_t* qSw1   = (uint32_t*)alloc((size_t)B_ * S_ * S_ * 2); // 16.8 MB
  uint32_t* qSw2   = (uint32_t*)alloc((size_t)B_ * S_ * S_ * 2); // 16.8 MB
  float* denomArr  = (float*)alloc(NR * 4);
  float* ZxtA      = (float*)alloc(NR * 4);
  float* Zrow      = (float*)alloc((size_t)B_ * S_ * 4);
  float* baseZ     = (float*)alloc((size_t)B_ * 4);
  float* accs      = (float*)alloc((3 * B_ + 1) * 4);
  float* accReg = accs;
  float* accOut = accs + B_;
  float* accNorm = accs + 2 * B_;
  float* nllSum = accs + 3 * B_;

  k_zero <<<1, 128, 0, stream>>>(accs, 3 * B_ + 1);
  k_rows <<<(unsigned)NR, 256, 0, stream>>>(qt0, rate, xlog, data, xt,
                                            aSig, aReg, denomArr, ZxtA, nllSum);
  k_zrow <<<B_, 512, 0, stream>>>(rate, Zrow);
  k_basez<<<B_, 256, 0, stream>>>(ZxtA, baseZ);
  dim3 gs(512, B_);
  k_swz  <<<gs, 256, 0, stream>>>(qt0, qSw1, qSw2);
  dim3 gm(31, B_);
  k_main <<<gm, 256, 0, stream>>>(qt0, data, aSig, aReg, qSw1, qSw2,
                                  denomArr, ZxtA, Zrow, baseZ,
                                  accReg, accOut, accNorm);
  k_final<<<1, 32, 0, stream>>>(accReg, accOut, accNorm, nllSum, out);
}
